// MultiHeadAttention_82652350644463
// MI455X (gfx1250) — matrix operand order repaired and checked by reading, not compiled, not run on hardware
//
#include <hip/hip_runtime.h>
#include <hip/hip_bf16.h>

typedef __attribute__((ext_vector_type(16))) _Float16     v16h;
typedef __attribute__((ext_vector_type(8)))  float        v8f;
typedef __attribute__((ext_vector_type(4)))  unsigned int u32x4;
typedef __attribute__((ext_vector_type(4)))  unsigned int su32x4;
typedef __attribute__((ext_vector_type(8)))  unsigned int su32x8;

union Frag16 { v16h v; u32x4 q[2]; };

#define DIM    768
#define HEADS  12
#define HD     64
#define SEQ    1024
#define BATCH  8
#define TOKENS (BATCH * SEQ)   // 8192
#define QKVN   (3 * DIM)       // 2304

__device__ __forceinline__ v8f wmma_f16(v16h a, v16h b, v8f c) {
  return __builtin_amdgcn_wmma_f32_16x16x32_f16(false, a, false, b, (short)0, c, false, false);
}

// --- Tensor Data Mover: 2D tile load, global f16 -> LDS, with row padding. ---
// D# group0 (128b) + group1 (256b) per CDNA5 ISA.
__device__ __forceinline__ void tdm_load_2d(unsigned lds_addr, const void* gsrc,
                                            unsigned tile_w, unsigned tile_h,
                                            unsigned row_stride_elems,
                                            unsigned pad_interval_code,
                                            unsigned pad_amount_code) {
  unsigned long long ga = (unsigned long long)(size_t)gsrc;
  su32x4 g0 = { 1u,                                   // count=1, user mode
                lds_addr,                              // lds_addr (bytes)
                (unsigned)ga,                          // global_addr[31:0]
                (unsigned)((ga >> 32) & 0x1FFFFFFull)  // global_addr[56:32]
                    | (2u << 30) };                    // type=2 ("image")
  su32x8 g1 = {
      (1u << 16)                                       // data_size = 2 bytes
          | (1u << 20)                                 // pad_enable
          | (pad_interval_code << 22)
          | (pad_amount_code << 25),
      (tile_w & 0xFFFFu) << 16,                        // tensor_dim0[15:0]
      ((tile_w >> 16) & 0xFFFFu) | ((tile_h & 0xFFFFu) << 16),  // dim0 hi | dim1 lo
      ((tile_h >> 16) & 0xFFFFu) | ((tile_w & 0xFFFFu) << 16),  // dim1 hi | tile_dim0
      (tile_h & 0xFFFFu),                              // tile_dim1 | tile_dim2=0
      row_stride_elems,                                // tensor_dim0_stride[31:0]
      0u, 0u };
  asm volatile("tensor_load_to_lds %0, %1" :: "s"(g0), "s"(g1) : "memory");
}

__global__ void f32_to_f16_kernel(const float* __restrict__ in,
                                  _Float16* __restrict__ out, int n) {
  int i = blockIdx.x * blockDim.x + threadIdx.x;
  if (i < n) out[i] = (_Float16)in[i];
}

// W[K,N] f32 -> Wt[N,K] f16
__global__ void transpose_w_kernel(const float* __restrict__ in,
                                   _Float16* __restrict__ out, int K, int N) {
  int i = blockIdx.x * blockDim.x + threadIdx.x;
  if (i < N * K) {
    int n = i / K, k = i - n * K;
    out[i] = (_Float16)in[(size_t)k * N + n];
  }
}

// qkv[b,n,2*DIM + h*HD + d] -> vt[((b*H+h)*HD + d)*SEQ + n]
__global__ void v_transpose_kernel(const _Float16* __restrict__ qkv,
                                   _Float16* __restrict__ vt) {
  int i = blockIdx.x * blockDim.x + threadIdx.x;
  if (i < TOKENS * DIM) {
    int n = i & (SEQ - 1);
    int r = i >> 10;                 // r = (b*HEADS + h)*HD + d
    int d = r & (HD - 1);
    int bh = r >> 6;                 // b*HEADS + h
    int h = bh % HEADS, b = bh / HEADS;
    vt[i] = qkv[(size_t)(b * SEQ + n) * QKVN + 2 * DIM + h * HD + d];
  }
}

// C[M,N] = A[M,K](f16) * BT[N,K](f16, pre-transposed) + bias[N]; tile 128x64.
// Double-buffered TDM pipeline: issue tile i+1, wait TENSORcnt<=2, compute i.
template <bool OUT_F16>
__global__ __launch_bounds__(256) void gemm_wmma_kernel(
    const _Float16* __restrict__ A, const _Float16* __restrict__ BT,
    const float* __restrict__ bias, void* __restrict__ Cout,
    int M, int N, int K) {
  __shared__ __align__(16) _Float16 sA[2][128][40];   // 32 halfs + 8 pad = 80B pitch
  __shared__ __align__(16) _Float16 sBT[2][64][40];

  const int tid  = threadIdx.x;
  const int lane = tid & 31;
  const int wid  = tid >> 5;
  const int wm   = wid >> 1, wn = wid & 1;
  const int g    = lane >> 4, ln = lane & 15;
  const int m0   = blockIdx.y * 128;
  const int n0   = blockIdx.x * 64;

  const unsigned ldsA[2] = { (unsigned)(size_t)&sA[0][0][0],
                             (unsigned)(size_t)&sA[1][0][0] };
  const unsigned ldsB[2] = { (unsigned)(size_t)&sBT[0][0][0],
                             (unsigned)(size_t)&sBT[1][0][0] };
  const _Float16* Abase = A + (size_t)m0 * K;
  const _Float16* Bbase = BT + (size_t)n0 * K;

  v8f acc[2][2];
  acc[0][0] = {}; acc[0][1] = {}; acc[1][0] = {}; acc[1][1] = {};

  const int NT = K / 32;
  if (wid == 0) {
    // 32-half rows (64B): pad_interval code 3 (16 DWORDs), pad 4 DWORDs -> code 3
    tdm_load_2d(ldsA[0], Abase, 32, 128, (unsigned)K, 3, 3);
    tdm_load_2d(ldsB[0], Bbase, 32, 64, (unsigned)K, 3, 3);
  }

  for (int i = 0; i < NT; ++i) {
    const int buf = i & 1;
    __syncthreads();                      // readers of buf^1 (iter i-1) are done
    if (wid == 0) {
      if (i + 1 < NT) {
        tdm_load_2d(ldsA[buf ^ 1], Abase + (i + 1) * 32, 32, 128, (unsigned)K, 3, 3);
        tdm_load_2d(ldsB[buf ^ 1], Bbase + (i + 1) * 32, 32, 64, (unsigned)K, 3, 3);
        __builtin_amdgcn_s_wait_tensorcnt(2);   // tile i landed; i+1 in flight
      } else {
        __builtin_amdgcn_s_wait_tensorcnt(0);
      }
    }
    __syncthreads();                      // buf data visible to all waves

    Frag16 fa[2], fb[2];
#pragma unroll
    for (int tm = 0; tm < 2; ++tm) {
      const _Float16* row = &sA[buf][wm * 32 + tm * 16 + ln][0];
      fa[tm].q[0] = *(const u32x4*)(row + g * 8);
      fa[tm].q[1] = *(const u32x4*)(row + 16 + g * 8);
    }
#pragma unroll
    for (int tn = 0; tn < 2; ++tn) {
      const _Float16* row = &sBT[buf][wn * 32 + tn * 16 + ln][0];
      fb[tn].q[0] = *(const u32x4*)(row + g * 8);
      fb[tn].q[1] = *(const u32x4*)(row + 16 + g * 8);
    }
    acc[0][0] = wmma_f16(fa[0].v, fb[0].v, acc[0][0]);
    acc[0][1] = wmma_f16(fa[0].v, fb[1].v, acc[0][1]);
    acc[1][0] = wmma_f16(fa[1].v, fb[0].v, acc[1][0]);
    acc[1][1] = wmma_f16(fa[1].v, fb[1].v, acc[1][1]);
  }

#pragma unroll
  for (int tm = 0; tm < 2; ++tm)
#pragma unroll
    for (int tn = 0; tn < 2; ++tn)
#pragma unroll
      for (int r = 0; r < 8; ++r) {
        int grow = m0 + wm * 32 + tm * 16 + g * 8 + r;
        int gcol = n0 + wn * 32 + tn * 16 + ln;
        float val = acc[tm][tn][r] + bias[gcol];
        if (OUT_F16)
          ((_Float16*)Cout)[(size_t)grow * N + gcol] = (_Float16)val;
        else
          ((float*)Cout)[(size_t)grow * N + gcol] = val;
      }
}

// Flash attention; K/V^T tiles in a double-buffered TDM pipeline.
__global__ __launch_bounds__(128) void attn_kernel(const _Float16* __restrict__ qkv,
                                                   const _Float16* __restrict__ vt,
                                                   _Float16* __restrict__ ctx) {
  __shared__ __align__(16) _Float16 sK[2][64][72];    // 64 halfs + 8 pad = 144B pitch
  __shared__ __align__(16) _Float16 sVT[2][64][72];
  __shared__ __align__(16) _Float16 sP[4][16][72];

  const int b = blockIdx.z, h = blockIdx.y;
  const int q0   = blockIdx.x * 64;
  const int tid  = threadIdx.x;
  const int lane = tid & 31, wave = tid >> 5;
  const int g    = lane >> 4, ln = lane & 15;

  const unsigned ldsK[2] = { (unsigned)(size_t)&sK[0][0][0],
                             (unsigned)(size_t)&sK[1][0][0] };
  const unsigned ldsV[2] = { (unsigned)(size_t)&sVT[0][0][0],
                             (unsigned)(size_t)&sVT[1][0][0] };
  const _Float16* kbase = qkv + (size_t)(b * SEQ) * QKVN + DIM + h * HD;
  const _Float16* vbase = vt + (size_t)(b * HEADS + h) * HD * SEQ;

  Frag16 aq[2];
  {
    const int qm = q0 + wave * 16 + ln;
    const _Float16* qrow = qkv + (size_t)(b * SEQ + qm) * QKVN + h * HD;
#pragma unroll
    for (int kk = 0; kk < 2; ++kk) {
      aq[kk].q[0] = *(const u32x4*)(qrow + kk * 32 + g * 8);
      aq[kk].q[1] = *(const u32x4*)(qrow + kk * 32 + 16 + g * 8);
    }
  }

  const v8f vzero = {};
  float mrow[8], lrow[8];
  v8f ot[4];
#pragma unroll
  for (int r = 0; r < 8; ++r) { mrow[r] = -1e30f; lrow[r] = 0.f; }
#pragma unroll
  for (int t = 0; t < 4; ++t) ot[t] = vzero;

  const int NT = SEQ / 64;
  if (wave == 0) {
    // 64-half rows (128B): pad_interval code 4 (32 DWORDs), pad 4 DWORDs -> code 3
    tdm_load_2d(ldsK[0], kbase, 64, 64, QKVN, 4, 3);
    tdm_load_2d(ldsV[0], vbase, 64, 64, SEQ, 4, 3);
  }

  for (int it = 0; it < NT; ++it) {
    const int buf = it & 1;
    __syncthreads();                      // readers of buf^1 (iter it-1) are done
    if (wave == 0) {
      if (it + 1 < NT) {
        const int nk = (it + 1) * 64;
        tdm_load_2d(ldsK[buf ^ 1], kbase + (size_t)nk * QKVN, 64, 64, QKVN, 4, 3);
        tdm_load_2d(ldsV[buf ^ 1], vbase + nk, 64, 64, SEQ, 4, 3);
        __builtin_amdgcn_s_wait_tensorcnt(2);   // tile it landed; it+1 in flight
      } else {
        __builtin_amdgcn_s_wait_tensorcnt(0);
      }
    }
    __syncthreads();

    v8f st[4];
#pragma unroll
    for (int t = 0; t < 4; ++t) st[t] = vzero;
#pragma unroll
    for (int kk = 0; kk < 2; ++kk) {
#pragma unroll
      for (int t = 0; t < 4; ++t) {
        Frag16 fk;
        const _Float16* row = &sK[buf][t * 16 + ln][0];
        fk.q[0] = *(const u32x4*)(row + kk * 32 + g * 8);
        fk.q[1] = *(const u32x4*)(row + kk * 32 + 16 + g * 8);
        st[t] = wmma_f16(aq[kk].v, fk.v, st[t]);
      }
    }

    float vm[8];
#pragma unroll
    for (int r = 0; r < 8; ++r) {
      st[0][r] *= 0.125f; st[1][r] *= 0.125f; st[2][r] *= 0.125f; st[3][r] *= 0.125f;
      vm[r] = fmaxf(fmaxf(st[0][r], st[1][r]), fmaxf(st[2][r], st[3][r]));
    }
#pragma unroll
    for (int off = 1; off <= 8; off <<= 1)
#pragma unroll
      for (int r = 0; r < 8; ++r)
        vm[r] = fmaxf(vm[r], __shfl_xor(vm[r], off, 32));

    float alpha[8], rs[8];
#pragma unroll
    for (int r = 0; r < 8; ++r) {
      float mn = fmaxf(mrow[r], vm[r]);
      alpha[r] = __expf(mrow[r] - mn);
      mrow[r]  = mn;
      rs[r]    = 0.f;
    }
#pragma unroll
    for (int t = 0; t < 4; ++t)
#pragma unroll
      for (int r = 0; r < 8; ++r) {
        float p = __expf(st[t][r] - mrow[r]);
        rs[r] += p;
        sP[wave][g * 8 + r][t * 16 + ln] = (_Float16)p;
      }
#pragma unroll
    for (int off = 1; off <= 8; off <<= 1)
#pragma unroll
      for (int r = 0; r < 8; ++r)
        rs[r] += __shfl_xor(rs[r], off, 32);
#pragma unroll
    for (int r = 0; r < 8; ++r)
      lrow[r] = lrow[r] * alpha[r] + rs[r];
#pragma unroll
    for (int t = 0; t < 4; ++t)
#pragma unroll
      for (int r = 0; r < 8; ++r)
        ot[t][r] *= alpha[r];

    __syncthreads();   // P stores visible before A-layout reload

    Frag16 pa[2];
    const _Float16* prow = &sP[wave][ln][0];
#pragma unroll
    for (int kk = 0; kk < 2; ++kk) {
      pa[kk].q[0] = *(const u32x4*)(prow + kk * 32 + g * 8);
      pa[kk].q[1] = *(const u32x4*)(prow + kk * 32 + 16 + g * 8);
    }
#pragma unroll
    for (int t = 0; t < 4; ++t) {
#pragma unroll
      for (int kk = 0; kk < 2; ++kk) {
        Frag16 fv;
        const _Float16* vrow = &sVT[buf][t * 16 + ln][0];
        fv.q[0] = *(const u32x4*)(vrow + kk * 32 + g * 8);
        fv.q[1] = *(const u32x4*)(vrow + kk * 32 + 16 + g * 8);
        ot[t] = wmma_f16(pa[kk].v, fv.v, ot[t]);
      }
    }
  }

  const int qbase = q0 + wave * 16 + g * 8;
#pragma unroll
  for (int t = 0; t < 4; ++t)
#pragma unroll
    for (int r = 0; r < 8; ++r) {
      float val = ot[t][r] / lrow[r];
      ctx[(size_t)(b * SEQ + qbase + r) * DIM + h * HD + t * 16 + ln] = (_Float16)val;
    }
}

extern "C" void kernel_launch(void* const* d_in, const int* in_sizes, int n_in,
                              void* d_out, int out_size, void* d_ws, size_t ws_size,
                              hipStream_t stream) {
  (void)in_sizes; (void)n_in; (void)out_size; (void)ws_size;
  const float* x      = (const float*)d_in[0];
  const float* W_qkv  = (const float*)d_in[1];
  const float* b_qkv  = (const float*)d_in[2];
  const float* W_proj = (const float*)d_in[3];
  const float* b_proj = (const float*)d_in[4];
  float* out = (float*)d_out;

  char* ws = (char*)d_ws;
  size_t off = 0;
  auto carve = [&](size_t bytes) -> char* {
    char* p = ws + off;
    off += (bytes + 255) & ~(size_t)255;
    return p;
  };
  _Float16* xh     = (_Float16*)carve((size_t)TOKENS * DIM * 2);   // 12.6 MB
  _Float16* wqkvt  = (_Float16*)carve((size_t)DIM * QKVN * 2);     //  3.5 MB  [N][K]
  _Float16* wprojt = (_Float16*)carve((size_t)DIM * DIM * 2);      //  1.2 MB  [N][K]
  _Float16* qkvh   = (_Float16*)carve((size_t)TOKENS * QKVN * 2);  // 37.7 MB
  _Float16* vth    = (_Float16*)carve((size_t)TOKENS * DIM * 2);   // 12.6 MB
  _Float16* ctxh   = (_Float16*)carve((size_t)TOKENS * DIM * 2);   // 12.6 MB

  const int nx = TOKENS * DIM, nq = DIM * QKVN, np = DIM * DIM;
  f32_to_f16_kernel<<<(nx + 255) / 256, 256, 0, stream>>>(x, xh, nx);
  transpose_w_kernel<<<(nq + 255) / 256, 256, 0, stream>>>(W_qkv, wqkvt, DIM, QKVN);
  transpose_w_kernel<<<(np + 255) / 256, 256, 0, stream>>>(W_proj, wprojt, DIM, DIM);

  gemm_wmma_kernel<true><<<dim3(QKVN / 64, TOKENS / 128), 256, 0, stream>>>(
      xh, wqkvt, b_qkv, qkvh, TOKENS, QKVN, DIM);

  v_transpose_kernel<<<(nx + 255) / 256, 256, 0, stream>>>(qkvh, vth);

  attn_kernel<<<dim3(SEQ / 64, HEADS, BATCH), 128, 0, stream>>>(qkvh, vth, ctxh);

  gemm_wmma_kernel<false><<<dim3(DIM / 64, TOKENS / 128), 256, 0, stream>>>(
      ctxh, wprojt, b_proj, out, TOKENS, DIM, DIM);
}
